// ResNet9_48189533061611
// MI455X (gfx1250) — matrix operand order repaired and checked by reading, not compiled, not run on hardware
//
#include <hip/hip_runtime.h>
#include <math.h>

#define PI2 6.283185307179586f

// ---------- WMMA types ----------
typedef __attribute__((ext_vector_type(16))) __bf16        bf16x16;
typedef __attribute__((ext_vector_type(8)))  float         f32x8;
typedef __attribute__((ext_vector_type(8)))  unsigned int  u32x8;

union BFU { bf16x16 v; u32x8 u; };

__device__ __forceinline__ unsigned short f2bf(float f) {
  unsigned int u = __float_as_uint(f);
  unsigned int r = u + 0x7FFFu + ((u >> 16) & 1u);   // round-to-nearest-even
  return (unsigned short)(r >> 16);
}

__device__ __forceinline__ bf16x16 bf_neg(bf16x16 a) {   // flip sign bits (co-executes with WMMA)
  BFU t; t.v = a;
  t.u = t.u ^ 0x80008000u;
  return t.v;
}

#define WMMA_BF16(A, Bm, C) \
  __builtin_amdgcn_wmma_f32_16x16x32_bf16(false, (A), false, (Bm), (short)0, (C), false, false)

// ============================================================
// Weight pre-pack: f32 [F][M][K] -> bf16 WMMA A-fragments.
// Fragment stream layout: frag(f,mt,kt,lane)[16 halves] contiguous:
//   index = (((f*MT + mt)*KT + kt)*32 + lane)*16 + h
// A-fragment half mapping (contiguous K, matches B operand):
//   M = mt*16 + (lane&15); K = kt*32 + (lane>>4)*16 + h
// ============================================================
__global__ void pack_wA_kernel(const float* __restrict__ W,
                               unsigned short* __restrict__ P,
                               int F, int M, int K)
{
  const int MT = M >> 4, KT = K >> 5;
  size_t total = (size_t)F * MT * KT * 32;
  size_t i = (size_t)blockIdx.x * blockDim.x + threadIdx.x;
  if (i >= total) return;
  const int lane = (int)(i & 31);
  size_t t = i >> 5;
  const int kt = (int)(t % KT); t /= KT;
  const int mt = (int)(t % MT);
  const int f  = (int)(t / MT);
  const int ml = lane & 15, hi = lane >> 4;

  const float* src = W + ((size_t)f * M + (size_t)(mt * 16 + ml)) * K + kt * 32;
  unsigned short v[16];
#pragma unroll
  for (int h = 0; h < 8; ++h) {
    v[h]     = f2bf(src[hi * 16 + h]);
    v[8 + h] = f2bf(src[hi * 16 + 8 + h]);
  }
  unsigned short* dst = P + i * 16;
#pragma unroll
  for (int h = 0; h < 16; ++h) dst[h] = v[h];
}

// ============================================================
// Complex per-frequency GEMM:  Y[f][n][m] = sum_k W[f][m][k] * X[f][n][k]
// A: packed bf16 fragment streams (Wr, Wi); X complex bf16 [F][N][K];
// Y complex f32 [F][N][M].  One wave per 32x32 (2x2-tile) output block.
// Incrementing stream pointers; unconditional speculative prefetch.
// ============================================================
__global__ void cgemm_wmma_kernel(const unsigned short* __restrict__ Awr,
                                  const unsigned short* __restrict__ Awi,
                                  const unsigned short* __restrict__ Xr,
                                  const unsigned short* __restrict__ Xi,
                                  float* __restrict__ Yr,
                                  float* __restrict__ Yi,
                                  int F, int M, int N, int K)
{
  const int MT = M >> 4, NT = N >> 4, KT = K >> 5;
  const int MB = MT >> 1, NB = NT >> 1;          // 2x2 tile blocks
  const long tiles = (long)F * MB * NB;
  const long wid = (long)blockIdx.x * (blockDim.x >> 5) + (threadIdx.x >> 5);
  if (wid >= tiles) return;                       // wave-uniform: EXEC all-1s for WMMA

  const int f  = (int)(wid / (MB * NB));
  const int r0 = (int)(wid % (MB * NB));
  const int mt = (r0 / NB) * 2, nt = (r0 % NB) * 2;
  const int lane = threadIdx.x & 31, ml = lane & 15, hi = lane >> 4;

  // packed A streams: frag(f,mt,kt,lane); per-kt stride = 32 lanes * 16 halves = 512
  const size_t a0off = (((size_t)f * MT + mt)     * KT * 32 + lane) * 16;
  const size_t a1off = (((size_t)f * MT + mt + 1) * KT * 32 + lane) * 16;
  const unsigned short* pa0r = Awr + a0off;
  const unsigned short* pa1r = Awr + a1off;
  const unsigned short* pa0i = Awi + a0off;
  const unsigned short* pa1i = Awi + a1off;

  // B rows: lane holds 16 contiguous K (K = kt*32 + hi*16 + h), N = nt*16 + ml
  const unsigned short* pb0r = Xr + ((size_t)f * N + (size_t)(nt * 16 + ml)) * K + hi * 16;
  const unsigned short* pb1r = pb0r + (size_t)16 * K;
  const unsigned short* pb0i = Xi + ((size_t)f * N + (size_t)(nt * 16 + ml)) * K + hi * 16;
  const unsigned short* pb1i = pb0i + (size_t)16 * K;

  f32x8 r00 = {}, r01 = {}, r10 = {}, r11 = {};
  f32x8 i00 = {}, i01 = {}, i10 = {}, i11 = {};

  for (int kt = 0; kt < KT; ++kt) {
    // speculative prefetch of next k-slab of the streamed A fragments
    __builtin_prefetch(pa0r + 512, 0, 1);        // gfx1250 global_prefetch_b8
    __builtin_prefetch(pa1r + 512, 0, 1);
    __builtin_prefetch(pa0i + 512, 0, 1);
    __builtin_prefetch(pa1i + 512, 0, 1);

    bf16x16 A0r = *(const bf16x16*)pa0r;
    bf16x16 A1r = *(const bf16x16*)pa1r;
    bf16x16 A0p = *(const bf16x16*)pa0i;
    bf16x16 A1p = *(const bf16x16*)pa1i;
    bf16x16 B0r = *(const bf16x16*)pb0r;
    bf16x16 B1r = *(const bf16x16*)pb1r;
    bf16x16 B0i = *(const bf16x16*)pb0i;
    bf16x16 B1i = *(const bf16x16*)pb1i;
    bf16x16 A0n = bf_neg(A0p);                   // -Wi via sign-bit XOR
    bf16x16 A1n = bf_neg(A1p);

    // Yr += Wr*Xr - Wi*Xi ; Yi += Wr*Xi + Wi*Xr
    r00 = WMMA_BF16(A0r, B0r, r00); r00 = WMMA_BF16(A0n, B0i, r00);
    i00 = WMMA_BF16(A0r, B0i, i00); i00 = WMMA_BF16(A0p, B0r, i00);
    r01 = WMMA_BF16(A0r, B1r, r01); r01 = WMMA_BF16(A0n, B1i, r01);
    i01 = WMMA_BF16(A0r, B1i, i01); i01 = WMMA_BF16(A0p, B1r, i01);
    r10 = WMMA_BF16(A1r, B0r, r10); r10 = WMMA_BF16(A1n, B0i, r10);
    i10 = WMMA_BF16(A1r, B0i, i10); i10 = WMMA_BF16(A1p, B0r, i10);
    r11 = WMMA_BF16(A1r, B1r, r11); r11 = WMMA_BF16(A1n, B1i, r11);
    i11 = WMMA_BF16(A1r, B1i, i11); i11 = WMMA_BF16(A1p, B1r, i11);

    pa0r += 512; pa1r += 512; pa0i += 512; pa1i += 512;
    pb0r += 32;  pb1r += 32;  pb0i += 32;  pb1i += 32;
  }

  // C/D layout: lane (N = lane%16), 8 contiguous M starting at (lane/16)*8
  const int n0 = nt * 16 + ml;
  const int m0 = mt * 16 + hi * 8;
  float* yr00 = Yr + ((size_t)f * N + n0) * M + m0;
  float* yi00 = Yi + ((size_t)f * N + n0) * M + m0;
  float* yr01 = yr00 + (size_t)16 * M;            // nt+1
  float* yi01 = yi00 + (size_t)16 * M;
#pragma unroll
  for (int r = 0; r < 8; ++r) {
    yr00[r] = r00[r];      yi00[r] = i00[r];
    yr00[16 + r] = r10[r]; yi00[16 + r] = i10[r]; // mt+1 (m0+16)
    yr01[r] = r01[r];      yi01[r] = i01[r];
    yr01[16 + r] = r11[r]; yi01[16 + r] = i11[r];
  }
}

// ============================================================
// rfft2: per (b,c) HxW real tile -> F = H*(W/2+1) complex bins,
// written bf16 to X[f][b][cmat] (channel-duplicated when Cmat > Cin).
// ============================================================
__global__ void rfft2_kernel(const float* __restrict__ x,
                             unsigned short* __restrict__ Xr,
                             unsigned short* __restrict__ Xi,
                             int B, int Cin, int H, int W, int Cmat)
{
  const int Wf = W / 2 + 1;
  const int b = blockIdx.x, c = blockIdx.y;
  extern __shared__ float lds[];
  float* tile = lds;               // H*W
  float* Zr = tile + H * W;        // H*Wf
  float* Zi = Zr + H * Wf;

  const float* src = x + ((size_t)b * Cin + c) * H * W;
  for (int i = threadIdx.x; i < H * W; i += blockDim.x) tile[i] = src[i];
  __syncthreads();

  for (int t = threadIdx.x; t < H * Wf; t += blockDim.x) {  // row rDFT
    int row = t / Wf, j = t % Wf;
    float sr = 0.f, si = 0.f;
    float w0 = -PI2 * (float)j / (float)W;
    for (int wv = 0; wv < W; ++wv) {
      float s, cn; __sincosf(w0 * wv, &s, &cn);
      float v = tile[row * W + wv];
      sr += v * cn; si += v * s;
    }
    Zr[t] = sr; Zi[t] = si;
  }
  __syncthreads();

  for (int t = threadIdx.x; t < H * Wf; t += blockDim.x) {  // column complex DFT
    int i = t / Wf, j = t % Wf;
    float sr = 0.f, si = 0.f;
    float w0 = -PI2 * (float)i / (float)H;
    for (int h = 0; h < H; ++h) {
      float s, cn; __sincosf(w0 * h, &s, &cn);
      float zr = Zr[h * Wf + j], zi = Zi[h * Wf + j];
      sr += zr * cn - zi * s;
      si += zr * s + zi * cn;
    }
    unsigned short ur = f2bf(sr), ui = f2bf(si);
    size_t base = ((size_t)t * B + b) * Cmat;
    Xr[base + c] = ur; Xi[base + c] = ui;
    if (Cmat > Cin) { Xr[base + c + Cin] = ur; Xi[base + c + Cin] = ui; }
  }
}

// ============================================================
// irfft2: Y[f][b][m] complex f32 -> real [B][M][H][W]
// ============================================================
__global__ void irfft2_kernel(const float* __restrict__ Yr,
                              const float* __restrict__ Yi,
                              float* __restrict__ out,
                              int B, int M, int H, int W)
{
  const int Wf = W / 2 + 1, F = H * Wf;
  const int b = blockIdx.x, o = blockIdx.y;
  extern __shared__ float lds[];
  float* Ar = lds;        // F
  float* Ai = Ar + F;
  float* Zr = Ai + F;     // H*Wf
  float* Zi = Zr + F;

  for (int t = threadIdx.x; t < F; t += blockDim.x) {
    size_t idx = ((size_t)t * B + b) * M + o;
    Ar[t] = Yr[idx]; Ai[t] = Yi[idx];
  }
  __syncthreads();

  for (int t = threadIdx.x; t < H * Wf; t += blockDim.x) {  // inverse column DFT
    int h = t / Wf, j = t % Wf;
    float sr = 0.f, si = 0.f;
    float w0 = PI2 * (float)h / (float)H;
    for (int i = 0; i < H; ++i) {
      float s, cn; __sincosf(w0 * i, &s, &cn);
      float ar = Ar[i * Wf + j], ai = Ai[i * Wf + j];
      sr += ar * cn - ai * s;
      si += ar * s + ai * cn;
    }
    Zr[t] = sr; Zi[t] = si;
  }
  __syncthreads();

  const float inv = 1.f / (float)(H * W);
  float* dst = out + ((size_t)b * M + o) * H * W;
  for (int t = threadIdx.x; t < H * W; t += blockDim.x) {   // hermitian row irfft
    int h = t / W, wv = t % W;
    float acc = Zr[h * Wf + 0];
    acc += ((wv & 1) ? -1.f : 1.f) * Zr[h * Wf + (W / 2)];
    float w0 = PI2 * (float)wv / (float)W;
    for (int j = 1; j < W / 2; ++j) {
      float s, cn; __sincosf(w0 * j, &s, &cn);
      acc += 2.f * (Zr[h * Wf + j] * cn - Zi[h * Wf + j] * s);
    }
    dst[t] = acc * inv;
  }
}

// ============================================================
// BatchNorm (training stats) + ReLU (+ optional residual after ReLU)
// ============================================================
__global__ void bn_stats_kernel(const float* __restrict__ x,
                                const float* __restrict__ g,
                                const float* __restrict__ bta,
                                float* __restrict__ scale,
                                float* __restrict__ shift,
                                int B, int C, int HW)
{
  const int c = blockIdx.x;
  float s = 0.f, s2 = 0.f;
  const int n = B * HW;
  for (int i = threadIdx.x; i < n; i += blockDim.x) {
    int b = i / HW, hw = i % HW;
    float v = x[((size_t)b * C + c) * HW + hw];
    s += v; s2 += v * v;
  }
  __shared__ float rs[256], rq[256];
  rs[threadIdx.x] = s; rq[threadIdx.x] = s2;
  __syncthreads();
  for (int st = 128; st > 0; st >>= 1) {
    if (threadIdx.x < st) { rs[threadIdx.x] += rs[threadIdx.x + st]; rq[threadIdx.x] += rq[threadIdx.x + st]; }
    __syncthreads();
  }
  if (threadIdx.x == 0) {
    float fn = (float)n;
    float mean = rs[0] / fn;
    float var  = rq[0] / fn - mean * mean;
    float sc = g[c] * rsqrtf(var + 1e-5f);
    scale[c] = sc;
    shift[c] = bta[c] - mean * sc;
  }
}

__global__ void bn_act_kernel(const float* __restrict__ x,
                              const float* __restrict__ scale,
                              const float* __restrict__ shift,
                              const float* __restrict__ res,
                              float* __restrict__ y,
                              int C, int HW, size_t total)
{
  size_t i = (size_t)blockIdx.x * blockDim.x + threadIdx.x;
  if (i >= total) return;
  int c = (int)((i / HW) % C);
  float v = fmaxf(0.f, x[i] * scale[c] + shift[c]);
  if (res) v += res[i];
  y[i] = v;
}

// ============================================================
// 2x2 maxpool, stride 2
// ============================================================
__global__ void maxpool2_kernel(const float* __restrict__ x, float* __restrict__ y,
                                int BC, int H, int W)
{
  const int Ho = H / 2, Wo = W / 2;
  size_t total = (size_t)BC * Ho * Wo;
  size_t i = (size_t)blockIdx.x * blockDim.x + threadIdx.x;
  if (i >= total) return;
  int wo = (int)(i % Wo);
  int ho = (int)((i / Wo) % Ho);
  size_t bc = i / ((size_t)Wo * Ho);
  const float* p = x + (bc * H + 2 * ho) * W + 2 * wo;
  y[i] = fmaxf(fmaxf(p[0], p[1]), fmaxf(p[W], p[W + 1]));
}

// ============================================================
// conv1: 3x3, pad 1, 3->64
// ============================================================
__global__ void conv3x3_kernel(const float* __restrict__ x, const float* __restrict__ w,
                               float* __restrict__ y, int B, int Cin, int Cout, int H, int W)
{
  size_t total = (size_t)B * Cout * H * W;
  size_t i = (size_t)blockIdx.x * blockDim.x + threadIdx.x;
  if (i >= total) return;
  int wv = (int)(i % W);
  int h  = (int)((i / W) % H);
  int o  = (int)((i / ((size_t)W * H)) % Cout);
  int b  = (int)(i / ((size_t)W * H * Cout));
  float acc = 0.f;
  for (int c = 0; c < Cin; ++c)
    for (int kh = 0; kh < 3; ++kh) {
      int hh = h + kh - 1;
      if (hh < 0 || hh >= H) continue;
      for (int kw = 0; kw < 3; ++kw) {
        int ww = wv + kw - 1;
        if (ww < 0 || ww >= W) continue;
        acc += x[(((size_t)b * Cin + c) * H + hh) * W + ww] *
               w[(((size_t)o * Cin + c) * 3 + kh) * 3 + kw];
      }
    }
  y[i] = acc;
}

// ============================================================
// FC: out[b][j] = fc_b[j] + sum_k act[b][k] * fc_w[j][k]
// ============================================================
__global__ void fc_kernel(const float* __restrict__ x, const float* __restrict__ w,
                          const float* __restrict__ bias, float* __restrict__ out,
                          int B, int Kf, int O)
{
  int i = blockIdx.x * blockDim.x + threadIdx.x;
  if (i >= B * O) return;
  int j = i % O, b = i / O;
  float acc = bias[j];
  const float* xp = x + (size_t)b * Kf;
  const float* wp = w + (size_t)j * Kf;
  for (int k = 0; k < Kf; ++k) acc += xp[k] * wp[k];
  out[i] = acc;
}

// ============================================================
// Host orchestration
// ============================================================
static void launch_bn(const float* x, const float* g, const float* b, const float* res,
                      float* y, float* scl, float* sht, int B, int C, int HW, hipStream_t s)
{
  bn_stats_kernel<<<C, 256, 0, s>>>(x, g, b, scl, sht, B, C, HW);
  size_t total = (size_t)B * C * HW;
  bn_act_kernel<<<(unsigned)((total + 255) / 256), 256, 0, s>>>(x, scl, sht, res, y, C, HW, total);
}

static void launch_fwconv(const float* in, float* out,
                          const float* wr, const float* wi,
                          unsigned short* Awr, unsigned short* Awi,
                          unsigned short* Xr, unsigned short* Xi,
                          float* Yr, float* Yi,
                          int B, int Cin, int Cmat, int H, hipStream_t s)
{
  const int W = H, Wf = W / 2 + 1, F = H * Wf;
  const int M = Cmat, N = B, K = Cmat;

  // forward FFT (spatial -> bf16 freq, with channel duplication)
  size_t sh1 = (size_t)(H * W + 2 * H * Wf) * sizeof(float);
  rfft2_kernel<<<dim3(B, Cin), 64, sh1, s>>>(in, Xr, Xi, B, Cin, H, W, Cmat);

  // weight pre-pack into WMMA A-fragment order (Wr, Wi)
  size_t pthreads = (size_t)F * (M >> 4) * (K >> 5) * 32;
  pack_wA_kernel<<<(unsigned)((pthreads + 255) / 256), 256, 0, s>>>(wr, Awr, F, M, K);
  pack_wA_kernel<<<(unsigned)((pthreads + 255) / 256), 256, 0, s>>>(wi, Awi, F, M, K);

  // complex WMMA GEMM, 2x2 tile blocks per wave
  long tiles = (long)F * (M / 32) * (N / 32);
  cgemm_wmma_kernel<<<(unsigned)((tiles + 7) / 8), 256, 0, s>>>(
      Awr, Awi, Xr, Xi, Yr, Yi, F, M, N, K);

  // inverse FFT back to spatial f32
  size_t sh2 = (size_t)(4 * H * Wf) * sizeof(float);
  irfft2_kernel<<<dim3(B, Cmat), 64, sh2, s>>>(Yr, Yi, out, B, Cmat, H, W);
}

extern "C" void kernel_launch(void* const* d_in, const int* in_sizes, int n_in,
                              void* d_out, int out_size, void* d_ws, size_t ws_size,
                              hipStream_t stream)
{
  (void)in_sizes; (void)n_in; (void)out_size; (void)ws_size;
  const int B = 256;

  // ---- inputs (setup_inputs dict order) ----
  const float* x       = (const float*)d_in[0];
  const float* conv1_w = (const float*)d_in[1];
  const float* conv2_wr = (const float*)d_in[2],  *conv2_wi = (const float*)d_in[3];
  const float* res1a_wr = (const float*)d_in[4],  *res1a_wi = (const float*)d_in[5];
  const float* res1b_wr = (const float*)d_in[6],  *res1b_wi = (const float*)d_in[7];
  const float* conv3_wr = (const float*)d_in[8],  *conv3_wi = (const float*)d_in[9];
  const float* conv4_wr = (const float*)d_in[10], *conv4_wi = (const float*)d_in[11];
  const float* res2a_wr = (const float*)d_in[12], *res2a_wi = (const float*)d_in[13];
  const float* res2b_wr = (const float*)d_in[14], *res2b_wi = (const float*)d_in[15];
  const float* bng[8], *bnb[8];
  for (int i = 0; i < 8; ++i) { bng[i] = (const float*)d_in[16 + 2 * i]; bnb[i] = (const float*)d_in[17 + 2 * i]; }
  const float* fc_w = (const float*)d_in[32];
  const float* fc_b = (const float*)d_in[33];
  float* out = (float*)d_out;

  // ---- workspace carve ----
  size_t off = 0;
  char* ws = (char*)d_ws;
  auto carve = [&](size_t bytes) -> void* {
    void* p = ws + off;
    off += (bytes + 255) & ~(size_t)255;
    return p;
  };
  const size_t SPE = (size_t)B * 128 * 32 * 32;     // largest spatial tensor (elems)
  const size_t FQE = (size_t)544 * B * 128;         // largest freq activation (elems)
  const size_t APE = (size_t)144 * 256 * 256;       // largest packed-weight stream (elems)
  float* sp0 = (float*)carve(SPE * 4);
  float* sp1 = (float*)carve(SPE * 4);
  float* sp2 = (float*)carve(SPE * 4);
  unsigned short* Xr = (unsigned short*)carve(FQE * 2);
  unsigned short* Xi = (unsigned short*)carve(FQE * 2);
  float* Yr = (float*)carve(FQE * 4);
  float* Yi = (float*)carve(FQE * 4);
  unsigned short* Awr = (unsigned short*)carve(APE * 2);
  unsigned short* Awi = (unsigned short*)carve(APE * 2);
  float* scl = (float*)carve(256 * 4);
  float* sht = (float*)carve(256 * 4);

  // ---- conv1 + bn1 ----
  {
    size_t total = (size_t)B * 64 * 32 * 32;
    conv3x3_kernel<<<(unsigned)((total + 255) / 256), 256, 0, stream>>>(x, conv1_w, sp0, B, 3, 64, 32, 32);
    launch_bn(sp0, bng[0], bnb[0], nullptr, sp1, scl, sht, B, 64, 1024, stream);
  }
  // ---- conv2 (64 -> 128, 32x32) + bn2 + pool -> sp2 [B,128,16,16] ----
  launch_fwconv(sp1, sp0, conv2_wr, conv2_wi, Awr, Awi, Xr, Xi, Yr, Yi, B, 64, 128, 32, stream);
  launch_bn(sp0, bng[1], bnb[1], nullptr, sp1, scl, sht, B, 128, 1024, stream);
  {
    size_t total = (size_t)B * 128 * 16 * 16;
    maxpool2_kernel<<<(unsigned)((total + 255) / 256), 256, 0, stream>>>(sp1, sp2, B * 128, 32, 32);
  }
  // ---- res1a: bn3(relu) -> sp1 ----
  launch_fwconv(sp2, sp0, res1a_wr, res1a_wi, Awr, Awi, Xr, Xi, Yr, Yi, B, 128, 128, 16, stream);
  launch_bn(sp0, bng[2], bnb[2], nullptr, sp1, scl, sht, B, 128, 256, stream);
  // ---- res1b: bn4(relu) + residual(sp2) -> sp1 ----
  launch_fwconv(sp1, sp0, res1b_wr, res1b_wi, Awr, Awi, Xr, Xi, Yr, Yi, B, 128, 128, 16, stream);
  launch_bn(sp0, bng[3], bnb[3], sp2, sp1, scl, sht, B, 128, 256, stream);
  // ---- conv3 (128 -> 256, 16x16) + bn5 + pool -> sp2 [B,256,8,8] ----
  launch_fwconv(sp1, sp0, conv3_wr, conv3_wi, Awr, Awi, Xr, Xi, Yr, Yi, B, 128, 256, 16, stream);
  launch_bn(sp0, bng[4], bnb[4], nullptr, sp1, scl, sht, B, 256, 256, stream);
  {
    size_t total = (size_t)B * 256 * 8 * 8;
    maxpool2_kernel<<<(unsigned)((total + 255) / 256), 256, 0, stream>>>(sp1, sp2, B * 256, 16, 16);
  }
  // ---- conv4 (256 -> 256, 8x8) + bn6 + pool -> sp2 [B,256,4,4] ----
  launch_fwconv(sp2, sp0, conv4_wr, conv4_wi, Awr, Awi, Xr, Xi, Yr, Yi, B, 256, 256, 8, stream);
  launch_bn(sp0, bng[5], bnb[5], nullptr, sp1, scl, sht, B, 256, 64, stream);
  {
    size_t total = (size_t)B * 256 * 4 * 4;
    maxpool2_kernel<<<(unsigned)((total + 255) / 256), 256, 0, stream>>>(sp1, sp2, B * 256, 8, 8);
  }
  // ---- res2a: bn7 -> sp1 ----
  launch_fwconv(sp2, sp0, res2a_wr, res2a_wi, Awr, Awi, Xr, Xi, Yr, Yi, B, 256, 256, 4, stream);
  launch_bn(sp0, bng[6], bnb[6], nullptr, sp1, scl, sht, B, 256, 16, stream);
  // ---- res2b: bn8 + residual(sp2) -> sp1 ----
  launch_fwconv(sp1, sp0, res2b_wr, res2b_wi, Awr, Awi, Xr, Xi, Yr, Yi, B, 256, 256, 4, stream);
  launch_bn(sp0, bng[7], bnb[7], sp2, sp1, scl, sht, B, 256, 16, stream);
  // ---- final pool [B,256,2,2] -> sp0, flatten, FC ----
  {
    size_t total = (size_t)B * 256 * 2 * 2;
    maxpool2_kernel<<<(unsigned)((total + 255) / 256), 256, 0, stream>>>(sp1, sp0, B * 256, 4, 4);
  }
  fc_kernel<<<(unsigned)((B * 10 + 255) / 256), 256, 0, stream>>>(sp0, fc_w, fc_b, out, B, 1024, 10);
}
